// HallowSurrogate_53489522704435
// MI455X (gfx1250) — matrix operand order repaired and checked by reading, not compiled, not run on hardware
//
#include <hip/hip_runtime.h>
#include <math.h>
#include <stdint.h>
#include <stddef.h>

// ---------------- problem constants ----------------
#define LATENT   24
#define N_STATE  70
#define N_KNOBS  4
#define N_CLIN   15
#define N_CRIT   33
#define OUTDIM   48
#define BATCH    32768
#define TSTEPS   32
#define SUBSTEPS 8

#define WAVES 8            // waves per workgroup in main kernel
#define ROWS_PER_WAVE 16

typedef __bf16 bf16_t;
typedef bf16_t v16bf __attribute__((ext_vector_type(16)));
typedef bf16_t v8bf  __attribute__((ext_vector_type(8)));
typedef float  v8f   __attribute__((ext_vector_type(8)));
typedef unsigned int u32x4 __attribute__((ext_vector_type(4)));
typedef int    i32x4 __attribute__((ext_vector_type(4)));
typedef int    i32x8 __attribute__((ext_vector_type(8)));

// ---- bf16 weight blob layout in d_ws (element offsets), W[n][kpad] transposed ----
#define W1T_OFF 0        // ode_w1t  192 x 32   (K real 28: 24 z + 4 knobs, pad->32)
#define W2T_OFF 6144     // ode_w2t  192 x 192
#define W3T_OFF 43008    // ode_w3t  192 x 192
#define W4T_OFF 79872    // ode_w4t   32 x 192  (N real 24, rows 24..31 zero)
#define D1T_OFF 86016    // dec_w1t  192 x 32   (K real 24, pad->32 zeros)
#define D2T_OFF 92160    // dec_w2t  128 x 192
#define D3T_OFF 116736   // dec_w3t   48 x 128
#define W_TOTAL 122880   // elements = 245,760 bytes = 12288 x 10 (TDM 2D tile)
#define Z0_BYTE_OFF 262144  // f32 z0 (B x 24) lives here in d_ws

// ---- bias LDS layout (float offsets) ----
#define OB1_OFF 0
#define OB2_OFF 192
#define OB3_OFF 384
#define OB4_OFF 576
#define DB1_OFF 600
#define DB2_OFF 792
#define DB3_OFF 920
#define BIAS_TOTAL 968

#if __has_builtin(__builtin_amdgcn_tensor_load_to_lds)
#define USE_TDM 1
#else
#define USE_TDM 0
#endif

// ---------------- fast branchless GELU ----------------
// bf16 activations (8-bit mantissa) make the exact-erf expansion pointless; the
// tanh form is within ~1e-3 of exact GELU. gfx1250 has V_TANH_F32 (1 trans op,
// co-executes with WMMA); fall back to a branchless exp-based tanh otherwise.
static __device__ __forceinline__ float tanh_fast(float u) {
#if __has_builtin(__builtin_amdgcn_tanhf)
  return __builtin_amdgcn_tanhf(u);
#else
  // tanh(u) = sign(u) * (1 - 2/(1+exp(2|u|))); exp->inf saturates to 1. Branchless.
  float au = fabsf(u);
  float e = __expf(2.0f * au);
  float t = 1.0f - 2.0f * __frcp_rn(1.0f + e);
  return copysignf(t, u);
#endif
}

static __device__ __forceinline__ float gelu_f(float x) {
  // 0.5*x*(1+tanh(sqrt(2/pi)*(x+0.044715*x^3)))
  float x2 = x * x;
  float u = 0.7978845608028654f * x * fmaf(0.044715f, x2, 1.0f);
  float hx = 0.5f * x;
  return fmaf(hx, tanh_fast(u), hx);
}

static __device__ __forceinline__ v8f vzero8() {
  v8f r;
#pragma unroll
  for (int i = 0; i < 8; ++i) r[i] = 0.0f;
  return r;
}

// per-wave LDS ordering fence (tiles are wave-private; LDS is in-order per wave,
// this just stops the compiler from reordering and drains DScnt)
static __device__ __forceinline__ void wave_sync() {
  __builtin_amdgcn_wave_barrier();
  asm volatile("s_wait_dscnt 0" ::: "memory");
  __builtin_amdgcn_wave_barrier();
}

static __device__ __forceinline__ v8f wmma_bf16(v16bf a, v16bf b, v8f c) {
  return __builtin_amdgcn_wmma_f32_16x16x32_bf16(false, a, false, b, (short)0, c,
                                                 false, false);
}

// A fragment: 16x32 bf16 tile, row-major act[m][k], ldk = row stride (elements).
// lane l: rows m=l&15; lanes 0-15 hold K {0..7,16..23}, lanes 16-31 {8..15,24..31}.
static __device__ __forceinline__ v16bf load_a(const bf16_t* act, int ldk, int kt,
                                               int lane) {
  int m  = lane & 15;
  int kb = (lane >> 4) << 3;  // 0 or 8
  const bf16_t* p = act + m * ldk + kt * 32;
  union { v16bf v; v8bf h[2]; } u;
  u.h[0] = *(const v8bf*)(p + kb);
  u.h[1] = *(const v8bf*)(p + 16 + kb);
  return u.v;
}

// B fragment: 32x16 bf16 tile from transposed weights wt[n][k], ldk = K stride.
// lane l: col n = ntile*16 + (l&15); lanes 0-15 hold K {0..7,16..23}, lanes 16-31 {8..15,24..31}.
static __device__ __forceinline__ v16bf load_b(const bf16_t* wt, int ldk, int ntile,
                                               int kt, int lane) {
  int n  = ntile * 16 + (lane & 15);
  int ks = kt * 32 + ((lane >> 4) << 3);  // 0 or 8
  const bf16_t* p = wt + n * ldk + ks;
  union { v16bf v; v8bf h[2]; } u;
  u.h[0] = *(const v8bf*)(p);
  u.h[1] = *(const v8bf*)(p + 16);
  return u.v;
}

// C/D accumulator: lane holds col n = (lane&15) [+16 for tile1], rows r + 8*(lane>>4).
static __device__ __forceinline__ void store_act_gelu(bf16_t* act, int ldk, int j,
                                                      int lane, v8f acc,
                                                      const float* bias) {
  int n = j * 16 + (lane & 15);
  int moff = (lane >> 4) << 3;
  float bb = bias[n];
#pragma unroll
  for (int r = 0; r < 8; ++r) {
    float v = gelu_f(acc[r] + bb);
    act[(moff + r) * ldk + n] = (bf16_t)v;
  }
}

// stage latent z (accumulator layout) into the 16x32 input tile, cols 0..23
static __device__ __forceinline__ void stage_z(bf16_t* tin, int lane, v8f z0v,
                                               v8f z1v) {
  int n0 = lane & 15;
  int moff = (lane >> 4) << 3;
#pragma unroll
  for (int r = 0; r < 8; ++r) tin[(moff + r) * 32 + n0] = (bf16_t)z0v[r];
  int n1 = 16 + n0;
  if (n1 < LATENT) {
#pragma unroll
    for (int r = 0; r < 8; ++r) tin[(moff + r) * 32 + n1] = (bf16_t)z1v[r];
  }
}

// ---------------- dz/dt = MLP([z, knobs]) ----------------
static __device__ __forceinline__ void ode_f(bf16_t* tin, bf16_t* th,
                                             const bf16_t* w, const float* bs,
                                             int lane, v8f zc0, v8f zc1, v8f& k0,
                                             v8f& k1) {
  stage_z(tin, lane, zc0, zc1);
  wave_sync();
  // layer 1: 32 -> 192 (single K tile); produce all 12 accums before activation
  {
    v16bf a = load_a(tin, 32, 0, lane);
    v8f acc[12];
#pragma unroll
    for (int j = 0; j < 12; ++j) {
      v16bf b = load_b(w + W1T_OFF, 32, j, 0, lane);
      acc[j] = vzero8();
      acc[j] = wmma_bf16(a, b, acc[j]);
    }
#pragma unroll
    for (int j = 0; j < 12; ++j)
      store_act_gelu(th, 192, j, lane, acc[j], bs + OB1_OFF);
  }
  wave_sync();
  // layers 2 & 3: 192 -> 192 (kept rolled so the 72-WMMA body appears once)
#pragma unroll 1
  for (int L = 0; L < 2; ++L) {
    const bf16_t* wt = w + (L == 0 ? W2T_OFF : W3T_OFF);
    const float* bb = bs + (L == 0 ? OB2_OFF : OB3_OFF);
    v8f acc[12];
#pragma unroll
    for (int j = 0; j < 12; ++j) acc[j] = vzero8();
#pragma unroll
    for (int kt = 0; kt < 6; ++kt) {
      v16bf a = load_a(th, 192, kt, lane);
#pragma unroll
      for (int j = 0; j < 12; ++j) {
        v16bf b = load_b(wt, 192, j, kt, lane);
        acc[j] = wmma_bf16(a, b, acc[j]);
      }
    }
    wave_sync();
#pragma unroll
    for (int j = 0; j < 12; ++j) store_act_gelu(th, 192, j, lane, acc[j], bb);
    wave_sync();
  }
  // layer 4: 192 -> 24 (2 n-tiles; tile1 cols >=24 are zero-weighted)
  {
    v8f a0 = vzero8(), a1 = vzero8();
#pragma unroll
    for (int kt = 0; kt < 6; ++kt) {
      v16bf a = load_a(th, 192, kt, lane);
      v16bf b0 = load_b(w + W4T_OFF, 192, 0, kt, lane);
      v16bf b1 = load_b(w + W4T_OFF, 192, 1, kt, lane);
      a0 = wmma_bf16(a, b0, a0);
      a1 = wmma_bf16(a, b1, a1);
    }
    int n0 = lane & 15;
    int n1 = 16 + n0;
    float bb0 = bs[OB4_OFF + n0];
    float bb1 = (n1 < LATENT) ? bs[OB4_OFF + n1] : 0.0f;
    k0 = a0 + bb0;
    k1 = a1 + bb1;
  }
}

// ---------------- decoder: z -> 192 -> 128 -> 48, scatter to d_out ----------------
static __device__ __forceinline__ void decode_store(bf16_t* tin, bf16_t* th,
                                                    const bf16_t* w, const float* bs,
                                                    int lane, v8f z0v, v8f z1v,
                                                    int ti, int row0,
                                                    float* __restrict__ out) {
  stage_z(tin, lane, z0v, z1v);  // knob cols 24..27 hit zero K-rows of dec_w1t
  wave_sync();
  // d1: 32 -> 192
  {
    v16bf a = load_a(tin, 32, 0, lane);
    v8f acc[12];
#pragma unroll
    for (int j = 0; j < 12; ++j) {
      v16bf b = load_b(w + D1T_OFF, 32, j, 0, lane);
      acc[j] = vzero8();
      acc[j] = wmma_bf16(a, b, acc[j]);
    }
#pragma unroll
    for (int j = 0; j < 12; ++j)
      store_act_gelu(th, 192, j, lane, acc[j], bs + DB1_OFF);
  }
  wave_sync();
  // d2: 192 -> 128
  {
    v8f acc[8];
#pragma unroll
    for (int j = 0; j < 8; ++j) acc[j] = vzero8();
#pragma unroll
    for (int kt = 0; kt < 6; ++kt) {
      v16bf a = load_a(th, 192, kt, lane);
#pragma unroll
      for (int j = 0; j < 8; ++j) {
        v16bf b = load_b(w + D2T_OFF, 192, j, kt, lane);
        acc[j] = wmma_bf16(a, b, acc[j]);
      }
    }
    wave_sync();
#pragma unroll
    for (int j = 0; j < 8; ++j) store_act_gelu(th, 192, j, lane, acc[j], bs + DB2_OFF);
    wave_sync();
  }
  // d3: 128 -> 48, write to global (tuple-split layout)
  {
    v8f o[3];
#pragma unroll
    for (int j = 0; j < 3; ++j) o[j] = vzero8();
#pragma unroll
    for (int kt = 0; kt < 4; ++kt) {
      v16bf a = load_a(th, 192, kt, lane);
#pragma unroll
      for (int j = 0; j < 3; ++j) {
        v16bf b = load_b(w + D3T_OFF, 128, j, kt, lane);
        o[j] = wmma_bf16(a, b, o[j]);
      }
    }
    int moff = (lane >> 4) << 3;
#pragma unroll
    for (int j = 0; j < 3; ++j) {
      int n = j * 16 + (lane & 15);
      float bb = bs[DB3_OFF + n];
#pragma unroll
      for (int r = 0; r < 8; ++r) {
        int row = row0 + moff + r;
        float v = o[j][r] + bb;
        size_t dst;
        if (n < N_CLIN)
          dst = ((size_t)ti * BATCH + row) * N_CLIN + n;
        else
          dst = (size_t)TSTEPS * BATCH * N_CLIN +
                ((size_t)ti * BATCH + row) * N_CRIT + (size_t)(n - N_CLIN);
        out[dst] = v;
      }
    }
  }
}

// ---------------- TDM: stage bf16 weight blob into LDS ----------------
#if USE_TDM
static __device__ __forceinline__ void tdm_load_weights(bf16_t* lds_dst,
                                                        const uint16_t* gsrc) {
  uint32_t lds_off = (uint32_t)(uintptr_t)lds_dst;
  uint64_t ga = (uint64_t)(uintptr_t)gsrc;
  u32x4 g0;
  g0[0] = 1u;                                   // count=1, user descriptor
  g0[1] = lds_off;                              // lds_addr (bytes)
  g0[2] = (uint32_t)(ga & 0xFFFFFFFFu);         // global_addr[31:0]
  g0[3] = (uint32_t)((ga >> 32) & 0x1FFFFFFu) | (2u << 30);  // addr[56:32], type=2
  const uint32_t dim0 = 12288u;                 // elements per row (2B each)
  const uint32_t rows = 10u;                    // 12288*10 = 122880 elements
  i32x8 g1;
  g1[0] = (int)(1u << 16);                                  // data_size=1 (2 bytes)
  g1[1] = (int)((dim0 & 0xFFFFu) << 16);                    // tensor_dim0 lo16
  g1[2] = (int)((dim0 >> 16) | ((rows & 0xFFFFu) << 16));   // dim0 hi / dim1 lo
  g1[3] = (int)((rows >> 16) | ((dim0 & 0xFFFFu) << 16));   // dim1 hi / tile_dim0
  g1[4] = (int)(rows & 0xFFFFu);                            // tile_dim1, tile_dim2=0
  g1[5] = (int)dim0;                                        // tensor_dim0_stride
  g1[6] = 0;
  g1[7] = 0;
  i32x4 zg = {0, 0, 0, 0};
#if defined(__clang_major__) && (__clang_major__ >= 23)
  i32x8 zg8 = {0, 0, 0, 0, 0, 0, 0, 0};
  __builtin_amdgcn_tensor_load_to_lds(g0, g1, zg, zg, zg8, 0);
#else
  __builtin_amdgcn_tensor_load_to_lds(g0, g1, zg, zg, 0);
#endif
#if __has_builtin(__builtin_amdgcn_s_wait_tensorcnt)
  __builtin_amdgcn_s_wait_tensorcnt(0);
#else
  asm volatile("s_wait_tensorcnt 0" ::: "memory");
#endif
}
#endif

// ---------------- main fused ODE + decoder kernel ----------------
__global__ __launch_bounds__(256, 1) void ode_main_kernel(
    const uint16_t* __restrict__ wsw, const float* __restrict__ z0ws,
    const float* __restrict__ knobs, const float* __restrict__ t_eval,
    const float* __restrict__ ob1, const float* __restrict__ ob2,
    const float* __restrict__ ob3, const float* __restrict__ ob4,
    const float* __restrict__ db1, const float* __restrict__ db2,
    const float* __restrict__ db3, float* __restrict__ out) {
  __shared__ __align__(256) bf16_t w_all[W_TOTAL];
  __shared__ __align__(256) float biasmem[BIAS_TOTAL];
  __shared__ __align__(256) bf16_t act_in[WAVES][16 * 32];
  __shared__ __align__(256) bf16_t act_h[WAVES][16 * 192];

  const int tid = (int)threadIdx.x;
  const int lane = tid & 31;
  const int wv = tid >> 5;
  const int row0 = ((int)blockIdx.x * WAVES + wv) * ROWS_PER_WAVE;

  // biases -> LDS
  for (int i = tid; i < 192; i += 256) {
    biasmem[OB1_OFF + i] = ob1[i];
    biasmem[OB2_OFF + i] = ob2[i];
    biasmem[OB3_OFF + i] = ob3[i];
    biasmem[DB1_OFF + i] = db1[i];
  }
  for (int i = tid; i < 128; i += 256) biasmem[DB2_OFF + i] = db2[i];
  for (int i = tid; i < 48; i += 256) biasmem[DB3_OFF + i] = db3[i];
  for (int i = tid; i < 24; i += 256) biasmem[OB4_OFF + i] = ob4[i];

  // weights -> LDS (single Tensor-Data-Mover op by wave 0)
#if USE_TDM
  if (wv == 0) tdm_load_weights(w_all, wsw);
#else
  {
    const bf16_t* g = (const bf16_t*)wsw;
    for (int i = tid; i < W_TOTAL; i += 256) w_all[i] = g[i];
  }
#endif
  __syncthreads();

  bf16_t* tin = &act_in[wv][0];
  bf16_t* th = &act_h[wv][0];

  // knobs into cols 24..27, zero pad cols 28..31 (NaN-safe padding)
  for (int idx = lane; idx < 16 * 8; idx += 32) {
    int r = idx >> 3, c = idx & 7;
    float v = (c < N_KNOBS) ? knobs[(size_t)(row0 + r) * N_KNOBS + c] : 0.0f;
    tin[r * 32 + 24 + c] = (bf16_t)v;
  }

  // z0 from encoder scratch (accumulator layout in registers)
  v8f z0v, z1v;
  {
    int n0 = lane & 15, n1 = 16 + (lane & 15);
    int moff = (lane >> 4) << 3;
#pragma unroll
    for (int r = 0; r < 8; ++r) {
      z0v[r] = z0ws[(size_t)(row0 + moff + r) * LATENT + n0];
      z1v[r] = (n1 < LATENT) ? z0ws[(size_t)(row0 + moff + r) * LATENT + n1] : 0.0f;
    }
  }
  wave_sync();

  const bf16_t* w = w_all;
  const float* bs = biasmem;

#pragma unroll 1
  for (int ti = 0; ti < TSTEPS; ++ti) {
    decode_store(tin, th, w, bs, lane, z0v, z1v, ti, row0, out);
    if (ti == TSTEPS - 1) break;
    float dt = t_eval[ti + 1] - t_eval[ti];
    float hstep = dt / (float)SUBSTEPS;
#pragma unroll 1
    for (int s = 0; s < SUBSTEPS; ++s) {
      v8f k0 = vzero8(), k1 = vzero8();
      v8f s0 = vzero8(), s1 = vzero8();
#pragma unroll 1
      for (int st = 0; st < 4; ++st) {
        float ci = (st == 0) ? 0.0f : ((st == 3) ? 1.0f : 0.5f);
        float cw = (st == 1 || st == 2) ? 2.0f : 1.0f;
        v8f zc0 = z0v + (hstep * ci) * k0;
        v8f zc1 = z1v + (hstep * ci) * k1;
        ode_f(tin, th, w, bs, lane, zc0, zc1, k0, k1);
        s0 += cw * k0;
        s1 += cw * k1;
      }
      z0v += (hstep / 6.0f) * s0;
      z1v += (hstep / 6.0f) * s1;
    }
  }
}

// ---------------- weight pack: f32 (fi,fo) -> bf16 transposed [nrows][kpad] ----------
__global__ void pack_kernel(const float* __restrict__ src, uint16_t* __restrict__ dst,
                            int fi, int fo, int kpad, int nrows) {
  int idx = (int)(blockIdx.x * blockDim.x + threadIdx.x);
  int total = nrows * kpad;
  if (idx >= total) return;
  int n = idx / kpad, k = idx % kpad;
  float v = (k < fi && n < fo) ? src[(size_t)k * fo + n] : 0.0f;
  union { bf16_t h; uint16_t u; } cv;
  cv.h = (bf16_t)v;
  dst[idx] = cv.u;
}

// ---------------- encoder: wave-per-row VALU (LayerNorms need row reductions) -------
static __device__ __forceinline__ float wred32(float v) {
#pragma unroll
  for (int m = 16; m > 0; m >>= 1) v += __shfl_xor(v, m, 32);
  return v;
}

__global__ __launch_bounds__(256) void encoder_kernel(
    const float* __restrict__ state, const float* __restrict__ knobs,
    const float* __restrict__ w1, const float* __restrict__ b1,
    const float* __restrict__ g1, const float* __restrict__ be1,
    const float* __restrict__ w2, const float* __restrict__ b2,
    const float* __restrict__ g2, const float* __restrict__ be2,
    const float* __restrict__ w3, const float* __restrict__ b3,
    float* __restrict__ z0out) {
  __shared__ float xs[8][80];
  __shared__ float hbuf[8][256];
  const int tid = (int)threadIdx.x, lane = tid & 31, wv = tid >> 5;
  const int row = (int)blockIdx.x * 8 + wv;

  for (int k = lane; k < 80; k += 32) {
    float v = 0.0f;
    if (k < N_STATE) v = state[(size_t)row * N_STATE + k];
    else if (k < N_STATE + N_KNOBS) v = knobs[(size_t)row * N_KNOBS + (k - N_STATE)];
    xs[wv][k] = v;
  }
  wave_sync();

  // layer 1: 74 -> 256, LN, GELU
  float a[8];
#pragma unroll
  for (int j = 0; j < 8; ++j) a[j] = b1[j * 32 + lane];
  for (int k = 0; k < N_STATE + N_KNOBS; ++k) {
    float xv = xs[wv][k];
#pragma unroll
    for (int j = 0; j < 8; ++j) a[j] += xv * w1[(size_t)k * 256 + j * 32 + lane];
  }
  {
    float s = 0.0f;
#pragma unroll
    for (int j = 0; j < 8; ++j) s += a[j];
    float mu = wred32(s) * (1.0f / 256.0f);
    float vs = 0.0f;
#pragma unroll
    for (int j = 0; j < 8; ++j) { float d = a[j] - mu; vs += d * d; }
    float rstd = rsqrtf(wred32(vs) * (1.0f / 256.0f) + 1e-5f);
#pragma unroll
    for (int j = 0; j < 8; ++j) {
      int n = j * 32 + lane;
      hbuf[wv][n] = gelu_f((a[j] - mu) * rstd * g1[n] + be1[n]);
    }
  }
  wave_sync();

  // layer 2: 256 -> 128, LN, GELU
  float a2[4];
#pragma unroll
  for (int j = 0; j < 4; ++j) a2[j] = b2[j * 32 + lane];
  for (int k = 0; k < 256; ++k) {
    float hv = hbuf[wv][k];
#pragma unroll
    for (int j = 0; j < 4; ++j) a2[j] += hv * w2[(size_t)k * 128 + j * 32 + lane];
  }
  {
    float s = 0.0f;
#pragma unroll
    for (int j = 0; j < 4; ++j) s += a2[j];
    float mu = wred32(s) * (1.0f / 128.0f);
    float vs = 0.0f;
#pragma unroll
    for (int j = 0; j < 4; ++j) { float d = a2[j] - mu; vs += d * d; }
    float rstd = rsqrtf(wred32(vs) * (1.0f / 128.0f) + 1e-5f);
    wave_sync();  // all reads of hbuf done before overwrite
#pragma unroll
    for (int j = 0; j < 4; ++j) {
      int n = j * 32 + lane;
      hbuf[wv][n] = gelu_f((a2[j] - mu) * rstd * g2[n] + be2[n]);
    }
  }
  wave_sync();

  // layer 3: 128 -> 24
  if (lane < LATENT) {
    float acc = b3[lane];
    for (int k = 0; k < 128; ++k) acc += hbuf[wv][k] * w3[(size_t)k * LATENT + lane];
    z0out[(size_t)row * LATENT + lane] = acc;
  }
}

// ---------------- launch ----------------
extern "C" void kernel_launch(void* const* d_in, const int* in_sizes, int n_in,
                              void* d_out, int out_size, void* d_ws, size_t ws_size,
                              hipStream_t stream) {
  (void)in_sizes; (void)n_in; (void)out_size; (void)ws_size;
  const float* state  = (const float*)d_in[0];
  const float* knobs  = (const float*)d_in[1];
  const float* t_eval = (const float*)d_in[2];
  const float* enc_w1 = (const float*)d_in[3];
  const float* enc_b1 = (const float*)d_in[4];
  const float* enc_g1 = (const float*)d_in[5];
  const float* enc_be1= (const float*)d_in[6];
  const float* enc_w2 = (const float*)d_in[7];
  const float* enc_b2 = (const float*)d_in[8];
  const float* enc_g2 = (const float*)d_in[9];
  const float* enc_be2= (const float*)d_in[10];
  const float* enc_w3 = (const float*)d_in[11];
  const float* enc_b3 = (const float*)d_in[12];
  const float* ode_w1 = (const float*)d_in[13];
  const float* ode_b1 = (const float*)d_in[14];
  const float* ode_w2 = (const float*)d_in[15];
  const float* ode_b2 = (const float*)d_in[16];
  const float* ode_w3 = (const float*)d_in[17];
  const float* ode_b3 = (const float*)d_in[18];
  const float* ode_w4 = (const float*)d_in[19];
  const float* ode_b4 = (const float*)d_in[20];
  const float* dec_w1 = (const float*)d_in[21];
  const float* dec_b1 = (const float*)d_in[22];
  const float* dec_w2 = (const float*)d_in[23];
  const float* dec_b2 = (const float*)d_in[24];
  const float* dec_w3 = (const float*)d_in[25];
  const float* dec_b3 = (const float*)d_in[26];

  uint16_t* wsw = (uint16_t*)d_ws;
  float* z0ws = (float*)((char*)d_ws + Z0_BYTE_OFF);

  struct PackArg { const float* src; int off, fi, fo, kpad, nrows; };
  const PackArg packs[7] = {
      {ode_w1, W1T_OFF, 28, 192, 32, 192},
      {ode_w2, W2T_OFF, 192, 192, 192, 192},
      {ode_w3, W3T_OFF, 192, 192, 192, 192},
      {ode_w4, W4T_OFF, 192, 24, 192, 32},
      {dec_w1, D1T_OFF, 24, 192, 32, 192},
      {dec_w2, D2T_OFF, 192, 128, 192, 128},
      {dec_w3, D3T_OFF, 128, 48, 128, 48},
  };
  for (int i = 0; i < 7; ++i) {
    int total = packs[i].nrows * packs[i].kpad;
    pack_kernel<<<(total + 255) / 256, 256, 0, stream>>>(
        packs[i].src, wsw + packs[i].off, packs[i].fi, packs[i].fo, packs[i].kpad,
        packs[i].nrows);
  }

  encoder_kernel<<<BATCH / 8, 256, 0, stream>>>(
      state, knobs, enc_w1, enc_b1, enc_g1, enc_be1, enc_w2, enc_b2, enc_g2,
      enc_be2, enc_w3, enc_b3, z0ws);

  ode_main_kernel<<<BATCH / (WAVES * ROWS_PER_WAVE), 256, 0, stream>>>(
      wsw, z0ws, knobs, t_eval, ode_b1, ode_b2, ode_b3, ode_b4, dec_b1, dec_b2,
      dec_b3, (float*)d_out);
}
